// LSTMregrModel_14723147891088
// MI455X (gfx1250) — compile-verified, matrix operand order checked by reading, not run on hardware
//
#include <hip/hip_runtime.h>

// Problem sizes (fixed by the reference)
#define B_   128
#define T_   512
#define D_   128
#define H_   1024
#define H4_  4096

typedef __attribute__((ext_vector_type(16))) __bf16 v16bf;
typedef __attribute__((ext_vector_type(8)))  float  v8f;

static __device__ __forceinline__ float sigmoidf_(float x) {
    return 1.0f / (1.0f + __expf(-x));
}

// ---------------------------------------------------------------------------
// Elementwise f32 -> bf16 conversion (weights + x), grid-stride
// ---------------------------------------------------------------------------
__global__ __launch_bounds__(256) void f32_to_bf16_k(const float* __restrict__ in,
                                                     __bf16* __restrict__ out, int n) {
    for (int i = blockIdx.x * blockDim.x + threadIdx.x; i < n;
         i += gridDim.x * blockDim.x)
        out[i] = (__bf16)in[i];
}

__global__ __launch_bounds__(256) void zero_u32_k(unsigned int* __restrict__ p, int n) {
    for (int i = blockIdx.x * blockDim.x + threadIdx.x; i < n;
         i += gridDim.x * blockDim.x)
        p[i] = 0u;
}

// ---------------------------------------------------------------------------
// One LSTM timestep:  g = x_t @ Wih^T + h_{t-1} @ Whh^T + bih + bhh
// Block = 128 threads = 4 waves. blockIdx.x -> batch tile (16 rows),
// blockIdx.y*4 + wave -> 16 hidden units. Each wave keeps 4 accumulators
// (i,f,g,o gate columns for its hidden units) so the cell update is local.
// Two K-phases (input proj, recurrent proj) reuse one 32KB LDS A-slab.
// ---------------------------------------------------------------------------
__global__ __launch_bounds__(128) void lstm_step_k(
    const __bf16* __restrict__ xbase,  // rows: xbase + m*xStride, length xK
    int xStride, int xK,
    const __bf16* __restrict__ hprev,  // [B,H] rows, stride H
    const __bf16* __restrict__ Wih,    // [4H, xK] row-major (bf16)
    const __bf16* __restrict__ Whh,    // [4H, H]  row-major (bf16)
    const float* __restrict__ bih, const float* __restrict__ bhh,
    float* __restrict__ c,             // [B,H] f32 cell state
    __bf16* __restrict__ hout)         // [B,H] slice of [T,B,H] sequence
{
    __shared__ __bf16 As[16 * H_];     // 32KB A slab (16 rows x up to 1024)

    const int tid  = threadIdx.x;
    const int lane = tid & 31;
    const int wave = tid >> 5;
    const int mbase = blockIdx.x * 16;
    const int jbase = (blockIdx.y * 4 + wave) * 16;
    const int nloc = lane & 15;
    const int half = lane >> 4;
    const int j = jbase + nloc;

    v8f acc[4] = {};                   // i, f, g, o accumulators (f32)

    #pragma unroll
    for (int phase = 0; phase < 2; ++phase) {
        const __bf16* Asrc   = phase ? hprev : xbase;
        const int     K      = phase ? H_    : xK;
        const int     stride = phase ? H_    : xStride;
        const __bf16* W      = phase ? Whh   : Wih;

        // Cooperative stage of 16 A rows into LDS (16B vector copies)
        const int k16 = K >> 3;        // # of 8-elem (16B) chunks per row
        for (int idx = tid; idx < 16 * k16; idx += 128) {
            const int row = idx / k16;
            const int k8  = (idx - row * k16) << 3;
            *(uint4*)(&As[row * K + k8]) =
                *(const uint4*)(Asrc + (size_t)(mbase + row) * stride + k8);
        }
        __syncthreads();

        for (int kb = 0; kb < K; kb += 32) {
            const int koff = kb + half * 16;
            v16bf a = *(const v16bf*)(&As[(lane & 15) * K + koff]);
            #pragma unroll
            for (int gt = 0; gt < 4; ++gt) {
                v16bf b = *(const v16bf*)(W + (size_t)(gt * H_ + j) * K + koff);
                acc[gt] = __builtin_amdgcn_wmma_f32_16x16x32_bf16(
                    false, a, false, b, (short)0, acc[gt], false, false);
            }
        }
        __syncthreads();
    }

    // Per-column gate biases
    const float bi = bih[0 * H_ + j] + bhh[0 * H_ + j];
    const float bfv = bih[1 * H_ + j] + bhh[1 * H_ + j];
    const float bg = bih[2 * H_ + j] + bhh[2 * H_ + j];
    const float bo = bih[3 * H_ + j] + bhh[3 * H_ + j];

    // C/D layout: VGPR r, lanes 0-15 -> M=r, lanes 16-31 -> M=8+r
    #pragma unroll
    for (int r = 0; r < 8; ++r) {
        const int m = mbase + half * 8 + r;
        const size_t off = (size_t)m * H_ + j;
        const float ig = sigmoidf_(acc[0][r] + bi);
        const float fg = sigmoidf_(acc[1][r] + bfv);
        const float gg = tanhf(acc[2][r] + bg);
        const float og = sigmoidf_(acc[3][r] + bo);
        const float cn = fg * c[off] + ig * gg;
        c[off] = cn;
        hout[off] = (__bf16)(og * tanhf(cn));
    }
}

// ---------------------------------------------------------------------------
// Attention scores: s[rid] = v . tanh(Wa h[rid] + ba), rid = t*B + b
// Block = 4 waves sharing one 16-row A slab; each wave covers 16 of 64
// output-column tiles, fuses tanh*v, reduces across lanes, atomically adds.
// ---------------------------------------------------------------------------
__global__ __launch_bounds__(128) void attn_scores_k(
    const __bf16* __restrict__ h1,   // [T*B, H]
    const __bf16* __restrict__ Wa,   // [H, H] row-major (B col n = row n)
    const float* __restrict__ ba,
    const float* __restrict__ v,
    float* __restrict__ scores)      // [T*B], pre-zeroed
{
    __shared__ __bf16 As[16 * H_];
    const int tid  = threadIdx.x;
    const int lane = tid & 31;
    const int wave = tid >> 5;
    const int nloc = lane & 15;
    const int half = lane >> 4;
    const int ridbase = blockIdx.x * 16;

    for (int idx = tid; idx < 16 * (H_ >> 3); idx += 128) {
        const int row = idx / (H_ >> 3);
        const int k8  = (idx - row * (H_ >> 3)) << 3;
        *(uint4*)(&As[row * H_ + k8]) =
            *(const uint4*)(h1 + (size_t)(ridbase + row) * H_ + k8);
    }
    __syncthreads();

    float part[8] = {0.f, 0.f, 0.f, 0.f, 0.f, 0.f, 0.f, 0.f};
    for (int nt = wave; nt < H_ / 16; nt += 4) {
        const int n = nt * 16 + nloc;
        v8f acc = {};
        for (int kb = 0; kb < H_; kb += 32) {
            const int koff = kb + half * 16;
            v16bf a = *(const v16bf*)(&As[(lane & 15) * H_ + koff]);
            v16bf b = *(const v16bf*)(Wa + (size_t)n * H_ + koff);
            acc = __builtin_amdgcn_wmma_f32_16x16x32_bf16(
                false, a, false, b, (short)0, acc, false, false);
        }
        const float ban = ba[n], vn = v[n];
        #pragma unroll
        for (int r = 0; r < 8; ++r)
            part[r] += tanhf(acc[r] + ban) * vn;
    }
    // Reduce over the 16 lanes of each half (columns of the tile)
    #pragma unroll
    for (int r = 0; r < 8; ++r) {
        float s = part[r];
        for (int off = 8; off; off >>= 1) s += __shfl_xor(s, off, 16);
        part[r] = s;
    }
    if (nloc == 0) {
        #pragma unroll
        for (int r = 0; r < 8; ++r)
            atomicAdd(&scores[ridbase + half * 8 + r], part[r]);
    }
}

// Softmax over T per batch row; scores/weights stored [T, B]
__global__ __launch_bounds__(256) void softmax_k(const float* __restrict__ scores,
                                                 float* __restrict__ w) {
    __shared__ float red[256];
    const int b = blockIdx.x, tid = threadIdx.x;
    const float s0 = scores[tid * B_ + b];
    const float s1 = scores[(tid + 256) * B_ + b];
    red[tid] = fmaxf(s0, s1);
    __syncthreads();
    for (int off = 128; off; off >>= 1) {
        if (tid < off) red[tid] = fmaxf(red[tid], red[tid + off]);
        __syncthreads();
    }
    const float m = red[0];
    __syncthreads();
    const float e0 = __expf(s0 - m), e1 = __expf(s1 - m);
    red[tid] = e0 + e1;
    __syncthreads();
    for (int off = 128; off; off >>= 1) {
        if (tid < off) red[tid] += red[tid + off];
        __syncthreads();
    }
    const float inv = 1.0f / red[0];
    w[tid * B_ + b] = e0 * inv;
    w[(tid + 256) * B_ + b] = e1 * inv;
}

// ctx[b,h] = sum_t w[t,b] * h1[t,b,h]
__global__ __launch_bounds__(256) void attn_ctx_k(const float* __restrict__ w,
                                                  const __bf16* __restrict__ h1,
                                                  float* __restrict__ ctx) {
    const int b = blockIdx.y;
    const int h = blockIdx.x * 256 + threadIdx.x;
    float s = 0.f;
    for (int t = 0; t < T_; ++t)
        s += w[t * B_ + b] * (float)h1[((size_t)t * B_ + b) * H_ + h];
    ctx[(size_t)b * H_ + h] = s;
}

// out[b] = ctx[b,:] . Wf[0,:] + bf[0]
__global__ __launch_bounds__(256) void out_head_k(const float* __restrict__ ctx,
                                                  const float* __restrict__ Wf,
                                                  const float* __restrict__ bf,
                                                  float* __restrict__ out) {
    __shared__ float red[256];
    const int b = blockIdx.x, tid = threadIdx.x;
    float s = 0.f;
    for (int h = tid; h < H_; h += 256)
        s += ctx[(size_t)b * H_ + h] * Wf[h];
    red[tid] = s;
    __syncthreads();
    for (int off = 128; off; off >>= 1) {
        if (tid < off) red[tid] += red[tid + off];
        __syncthreads();
    }
    if (tid == 0) out[b] = red[0] + bf[0];
}

// ---------------------------------------------------------------------------
extern "C" void kernel_launch(void* const* d_in, const int* in_sizes, int n_in,
                              void* d_out, int out_size, void* d_ws, size_t ws_size,
                              hipStream_t stream) {
    (void)in_sizes; (void)n_in; (void)out_size;

    const float* x    = (const float*)d_in[0];
    const float* Wih0 = (const float*)d_in[1];
    const float* Whh0 = (const float*)d_in[2];
    const float* bih0 = (const float*)d_in[3];
    const float* bhh0 = (const float*)d_in[4];
    const float* Wih1 = (const float*)d_in[5];
    const float* Whh1 = (const float*)d_in[6];
    const float* bih1 = (const float*)d_in[7];
    const float* bhh1 = (const float*)d_in[8];
    const float* Wa   = (const float*)d_in[9];
    const float* ba   = (const float*)d_in[10];
    const float* v    = (const float*)d_in[11];
    const float* Wf   = (const float*)d_in[12];
    const float* bf   = (const float*)d_in[13];

    // Workspace carving (256B aligned)
    char* p = (char*)d_ws;
    auto carve = [&](size_t bytes) -> char* {
        char* q = p;
        p += (bytes + 255) & ~(size_t)255;
        return q;
    };
    __bf16* xbf   = (__bf16*)carve((size_t)B_ * T_ * D_ * 2);
    __bf16* Wih0b = (__bf16*)carve((size_t)H4_ * D_ * 2);
    __bf16* Whh0b = (__bf16*)carve((size_t)H4_ * H_ * 2);
    __bf16* Wih1b = (__bf16*)carve((size_t)H4_ * H_ * 2);
    __bf16* Whh1b = (__bf16*)carve((size_t)H4_ * H_ * 2);
    __bf16* Wab   = (__bf16*)carve((size_t)H_ * H_ * 2);
    __bf16* h0seq = (__bf16*)carve((size_t)T_ * B_ * H_ * 2);  // [T,B,H]
    __bf16* h1seq = (__bf16*)carve((size_t)T_ * B_ * H_ * 2);  // [T,B,H]
    __bf16* hzero = (__bf16*)carve((size_t)B_ * H_ * 2);
    float*  cbuf  = (float*)carve((size_t)B_ * H_ * 4);
    float*  scores= (float*)carve((size_t)T_ * B_ * 4);
    float*  wsoft = (float*)carve((size_t)T_ * B_ * 4);
    float*  ctx   = (float*)carve((size_t)B_ * H_ * 4);
    if ((size_t)(p - (char*)d_ws) > ws_size) return;  // workspace too small

    // Precision staging: f32 -> bf16
    f32_to_bf16_k<<<1024, 256, 0, stream>>>(x,    xbf,   B_ * T_ * D_);
    f32_to_bf16_k<<<256,  256, 0, stream>>>(Wih0, Wih0b, H4_ * D_);
    f32_to_bf16_k<<<1024, 256, 0, stream>>>(Whh0, Whh0b, H4_ * H_);
    f32_to_bf16_k<<<1024, 256, 0, stream>>>(Wih1, Wih1b, H4_ * H_);
    f32_to_bf16_k<<<1024, 256, 0, stream>>>(Whh1, Whh1b, H4_ * H_);
    f32_to_bf16_k<<<256,  256, 0, stream>>>(Wa,   Wab,   H_ * H_);
    zero_u32_k<<<256, 256, 0, stream>>>((unsigned int*)hzero, B_ * H_ / 2);
    zero_u32_k<<<256, 256, 0, stream>>>((unsigned int*)cbuf,  B_ * H_);

    const dim3 sgrid(B_ / 16, H_ / 64);   // 8 batch tiles x 16 j-groups (4 waves each)

    // Layer 0: A = [x_t (K=128) | h_{t-1} (K=1024)]
    for (int t = 0; t < T_; ++t) {
        const __bf16* hp = t ? h0seq + (size_t)(t - 1) * B_ * H_ : hzero;
        lstm_step_k<<<sgrid, 128, 0, stream>>>(
            xbf + (size_t)t * D_, T_ * D_, D_,
            hp, Wih0b, Whh0b, bih0, bhh0, cbuf,
            h0seq + (size_t)t * B_ * H_);
    }

    // Layer 1: A = [h0_t (K=1024) | h_{t-1} (K=1024)]
    zero_u32_k<<<256, 256, 0, stream>>>((unsigned int*)cbuf, B_ * H_);
    for (int t = 0; t < T_; ++t) {
        const __bf16* hp = t ? h1seq + (size_t)(t - 1) * B_ * H_ : hzero;
        lstm_step_k<<<sgrid, 128, 0, stream>>>(
            h0seq + (size_t)t * B_ * H_, H_, H_,
            hp, Wih1b, Whh1b, bih1, bhh1, cbuf,
            h1seq + (size_t)t * B_ * H_);
    }

    // Attention + head
    zero_u32_k<<<256, 256, 0, stream>>>((unsigned int*)scores, T_ * B_);
    attn_scores_k<<<T_ * B_ / 16, 128, 0, stream>>>(h1seq, Wab, ba, v, scores);
    softmax_k<<<B_, 256, 0, stream>>>(scores, wsoft);
    attn_ctx_k<<<dim3(H_ / 256, B_), 256, 0, stream>>>(wsoft, h1seq, ctx);
    out_head_k<<<B_, 256, 0, stream>>>(ctx, Wf, bf, (float*)d_out);
}
